// SingleHeadAttention_9079560864368
// MI455X (gfx1250) — matrix operand order repaired and checked by reading, not compiled, not run on hardware
//
#include <hip/hip_runtime.h>

#define D_MODEL 1024
#define D_HEAD  64
#define S_LEN   4096
#define BATCH   4

typedef __attribute__((ext_vector_type(16))) _Float16 v16h;
typedef __attribute__((ext_vector_type(8)))  _Float16 v8h;
typedef __attribute__((ext_vector_type(8)))  float    v8f;
typedef __attribute__((ext_vector_type(4)))  float    v4f;

union H16 { v16h v; v8h h[2]; };

// D = A(16x32 f16) * B(32x16 f16) + C(16x16 f32)
static __device__ inline v8f wmma16(const H16& a, const H16& b, v8f c) {
  return __builtin_amdgcn_wmma_f32_16x16x32_f16(
      /*neg_a=*/false, a.v, /*neg_b=*/false, b.v,
      /*c_mod=*/(short)0, c, /*reuse_a=*/false, /*reuse_b=*/false);
}

// gfx1250 async copy: 16B global -> LDS per enabled lane, tracked by ASYNCcnt
static __device__ inline void async_ld_b128(unsigned lds_off, const void* gaddr) {
  asm volatile("global_load_async_to_lds_b128 %0, %1, off"
               :: "v"(lds_off), "v"(gaddr) : "memory");
}
static __device__ inline void wait_asynccnt0() {
  asm volatile("s_wait_asynccnt 0x0" ::: "memory");
}
// low 32 bits of a generic pointer to LDS == LDS byte address
static __device__ inline unsigned lds_addr(const void* p) {
  return (unsigned)(unsigned long long)p;
}

// load 8 consecutive f32 (16B-aligned) and convert to 8 f16
static __device__ inline v8h cvt8h(const float* __restrict__ p) {
  v4f a = *(const v4f*)p;
  v4f b = *(const v4f*)(p + 4);
  v8h r;
  r[0] = (_Float16)a[0]; r[1] = (_Float16)a[1];
  r[2] = (_Float16)a[2]; r[3] = (_Float16)a[3];
  r[4] = (_Float16)b[0]; r[5] = (_Float16)b[1];
  r[6] = (_Float16)b[2]; r[7] = (_Float16)b[3];
  return r;
}

// reductions across each 16-lane half (xor masks <16 never cross halves)
static __device__ inline float red_max16(float x) {
  x = fmaxf(x, __shfl_xor(x, 1));
  x = fmaxf(x, __shfl_xor(x, 2));
  x = fmaxf(x, __shfl_xor(x, 4));
  x = fmaxf(x, __shfl_xor(x, 8));
  return x;
}
static __device__ inline float red_sum16(float x) {
  x += __shfl_xor(x, 1);
  x += __shfl_xor(x, 2);
  x += __shfl_xor(x, 4);
  x += __shfl_xor(x, 8);
  return x;
}

// ---------------------------------------------------------------------------
// Fused Q/K/V projection: out[s,h] = sum_d X[s,d]*W[h,d] + b[h]
// One wave -> 16 rows x 64 cols. qh/kh row-major [B,S,64] f16, vh transposed
// [B,64,S] f16 so attention's V B-operand loads are contiguous.
// ---------------------------------------------------------------------------
__global__ void __launch_bounds__(256)
proj_kernel(const float* __restrict__ q, const float* __restrict__ k,
            const float* __restrict__ v,
            const float* __restrict__ WQ, const float* __restrict__ bQ,
            const float* __restrict__ WK, const float* __restrict__ bK,
            const float* __restrict__ WV, const float* __restrict__ bV,
            _Float16* __restrict__ qh, _Float16* __restrict__ kh,
            _Float16* __restrict__ vhT) {
  const int wave  = (blockIdx.x * blockDim.x + threadIdx.x) >> 5;
  const int lane  = threadIdx.x & 31;
  const int lm    = lane & 15;      // M row (A) / N col (B,C)
  const int hi    = lane >> 4;      // half-wave selector

  const int which = wave >> 10;     // 0=q,1=k,2=v  (1024 row-tiles each)
  const int rt    = wave & 1023;
  const int b     = rt >> 8;        // 256 tiles per batch
  const int s0    = (rt & 255) << 4;

  const float* X; const float* W; const float* bias;
  if (which == 0)      { X = q; W = WQ; bias = bQ; }
  else if (which == 1) { X = k; W = WK; bias = bK; }
  else                 { X = v; W = WV; bias = bV; }

  const float* xrow = X + (size_t)(b * S_LEN + s0 + lm) * D_MODEL;

  v8f acc[4] = {};                  // 4 N-tiles of 16 cols

  for (int k0 = 0; k0 < D_MODEL; k0 += 32) {
    // A fragment: lane<16 -> K {0..15}; lane>=16 -> K {16..31}
    H16 A;
    const int klo = k0 + hi * 16;
    A.h[0] = cvt8h(xrow + klo);
    A.h[1] = cvt8h(xrow + klo + 8);
#pragma unroll
    for (int t = 0; t < 4; ++t) {
      // B fragment: lane j holds col h = 16t + j%16, K = (j<16?0:16)+i
      const float* wrow = W + (size_t)(t * 16 + lm) * D_MODEL + k0 + hi * 16;
      H16 B;
      B.h[0] = cvt8h(wrow);
      B.h[1] = cvt8h(wrow + 8);
      acc[t] = wmma16(A, B, acc[t]);
    }
  }

  if (which != 2) {
    _Float16* out = (which == 0) ? qh : kh;
#pragma unroll
    for (int t = 0; t < 4; ++t) {
      const float bb = bias[t * 16 + lm];
#pragma unroll
      for (int r = 0; r < 8; ++r) {
        const int row = r + 8 * hi;             // C layout: M = r + 8*(lane>=16)
        out[(size_t)(b * S_LEN + s0 + row) * D_HEAD + t * 16 + lm] =
            (_Float16)(acc[t][r] + bb);
      }
    }
  } else {
    // transposed store: fixed h per lane, rows r consecutive in s -> b128
#pragma unroll
    for (int t = 0; t < 4; ++t) {
      const int h = t * 16 + lm;
      const float bb = bias[h];
      v8h pack;
#pragma unroll
      for (int r = 0; r < 8; ++r) pack[r] = (_Float16)(acc[t][r] + bb);
      *(v8h*)(vhT + (size_t)(b * D_HEAD + h) * S_LEN + s0 + 8 * hi) = pack;
    }
  }
}

// ---------------------------------------------------------------------------
// Causal flash attention, block-cooperative:
//  - 1 block = 8 waves = 128 consecutive queries of one batch
//  - 64-key K/V tiles double-buffered in LDS via async global->LDS copies
//    (ASYNCcnt + s_wait_asynccnt + one barrier per tile)
//  - per-wave online softmax; P relayout (C->A) through a private LDS tile
// ---------------------------------------------------------------------------
__global__ void __launch_bounds__(256)
attn_kernel(const _Float16* __restrict__ qh, const _Float16* __restrict__ kh,
            const _Float16* __restrict__ vhT, float* __restrict__ out) {
  __shared__ _Float16 ldsK[2][64 * 64];     // [buf][key][h]   8KB each
  __shared__ _Float16 ldsV[2][64 * 64];     // [buf][h][key]   8KB each
  __shared__ _Float16 ldsP[8][16 * 32];     // per-wave P tile 1KB

  const int tid  = threadIdx.x;
  const int wInB = tid >> 5;
  const int lane = tid & 31;
  const int lm   = lane & 15;
  const int hiw  = lane >> 4;

  const int blk = blockIdx.x;               // 32 blocks per batch (4096/128)
  const int b   = blk >> 5;
  const int q0b = (blk & 31) << 7;          // first query row of block
  const int q0  = q0b + wInB * 16;          // this wave's query tile

  const _Float16* kbase = kh  + (size_t)b * S_LEN * D_HEAD;   // [S][64]
  const _Float16* vbase = vhT + (size_t)b * D_HEAD * S_LEN;   // [64][S]

  // stage one 64-key block into buffer `buf` (all 256 threads cooperate;
  // 4 async b128 transfers per thread = 16KB total)
  auto stage = [&](int kb, int buf) {
    // K tile: rows kb..kb+63 of [S][64] are 8KB contiguous
    {
      const char* g = (const char*)(kbase + (size_t)kb * D_HEAD);
      unsigned l = lds_addr(&ldsK[buf][0]);
      async_ld_b128(l + tid * 16,        g + tid * 16);
      async_ld_b128(l + tid * 16 + 4096, g + tid * 16 + 4096);
    }
    // V tile: 64 rows (h) x 128B segments, global row stride S*2 bytes
    {
      unsigned l = lds_addr(&ldsV[buf][0]);
#pragma unroll
      for (int i = 0; i < 2; ++i) {
        const int c = tid + i * 256;                 // chunk id 0..511
        const int r = c >> 3;                        // h row
        const int o = (c & 7) * 8;                   // halves within row
        const char* g = (const char*)(vbase + (size_t)r * S_LEN + kb + o);
        async_ld_b128(l + c * 16, g);
      }
    }
  };

  // Q A-fragments for K(head dim)=0..31 and 32..63, loaded once
  const _Float16* qrow = qh + (size_t)(b * S_LEN + q0 + lm) * D_HEAD;
  const int klo = hiw * 16;
  H16 A0, A1;
  A0.h[0] = *(const v8h*)(qrow + klo);
  A0.h[1] = *(const v8h*)(qrow + klo + 8);
  A1.h[0] = *(const v8h*)(qrow + 32 + klo);
  A1.h[1] = *(const v8h*)(qrow + 32 + klo + 8);

  float mi[8], li[8];
  v8f acc[4] = {};
#pragma unroll
  for (int r = 0; r < 8; ++r) { mi[r] = -3.0e38f; li[r] = 0.f; }

  const float scale = 0.125f;               // 1/sqrt(64)
  _Float16* P = ldsP[wInB];
  const int kmaxw = q0 + 16;                // this wave's causal bound
  const int nblk  = (q0b >> 6) + 2;         // 64-key blocks for whole WG

  stage(0, 0);

  for (int i = 0; i < nblk; ++i) {
    wait_asynccnt0();                       // own async transfers landed
    __syncthreads();                        // everyone's landed; prev compute done
    if (i + 1 < nblk) stage((i + 1) * 64, (i + 1) & 1);

    const _Float16* Kt = ldsK[i & 1];
    const _Float16* Vt = ldsV[i & 1];
    const int kb = i * 64;

#pragma unroll
    for (int h32 = 0; h32 < 64; h32 += 32) {
      const int kb32 = kb + h32;
      if (kb32 < kmaxw) {                   // wave-uniform causal skip
        // ---- scores: 2 key sub-tiles x 2 K-steps, alternated accums ----
        v8f c0 = {}, c1 = {};
        const _Float16* kr0 = Kt + (h32 + lm) * D_HEAD;      // keys kb32+0..15
        const _Float16* kr1 = kr0 + 16 * D_HEAD;             // keys kb32+16..31
        H16 B0, B1;
        B0.h[0] = *(const v8h*)(kr0 + hiw * 16);
        B0.h[1] = *(const v8h*)(kr0 + hiw * 16 + 8);
        B1.h[0] = *(const v8h*)(kr1 + hiw * 16);
        B1.h[1] = *(const v8h*)(kr1 + hiw * 16 + 8);
        c0 = wmma16(A0, B0, c0);
        c1 = wmma16(A0, B1, c1);
        B0.h[0] = *(const v8h*)(kr0 + 32 + hiw * 16);
        B0.h[1] = *(const v8h*)(kr0 + 32 + hiw * 16 + 8);
        B1.h[0] = *(const v8h*)(kr1 + 32 + hiw * 16);
        B1.h[1] = *(const v8h*)(kr1 + 32 + hiw * 16 + 8);
        c0 = wmma16(A1, B0, c0);
        c1 = wmma16(A1, B1, c1);

        // ---- mask + online softmax (row = reg + 8*half, col = lane%16) ----
#pragma unroll
        for (int r = 0; r < 8; ++r) {
          const int row  = q0 + r + 8 * hiw;
          const int col0 = kb32 + lm;
          const float s0 = (col0 > row)      ? -1.0e30f : c0[r] * scale;
          const float s1 = (col0 + 16 > row) ? -1.0e30f : c1[r] * scale;
          const float bm   = red_max16(fmaxf(s0, s1));
          const float mnew = fmaxf(mi[r], bm);
          const float al   = __expf(mi[r] - mnew);
          const float p0   = __expf(s0 - mnew);
          const float p1   = __expf(s1 - mnew);
          li[r] = li[r] * al + red_sum16(p0 + p1);
          mi[r] = mnew;
#pragma unroll
          for (int t = 0; t < 4; ++t) acc[t][r] *= al;
          const int prow = r + 8 * hiw;
          P[prow * 32 + lm]      = (_Float16)p0;
          P[prow * 32 + lm + 16] = (_Float16)p1;
        }

        // ---- re-read P in A layout (16x32, K = 32 keys) ----
        H16 Ap;
        Ap.h[0] = *(const v8h*)(P + lm * 32 + klo);
        Ap.h[1] = *(const v8h*)(P + lm * 32 + klo + 8);

        // ---- P @ V: 4 head-dim N-tiles ----
#pragma unroll
        for (int t = 0; t < 4; ++t) {
          const _Float16* vr = Vt + (t * 16 + lm) * 64 + h32 + hiw * 16;
          H16 Bv;
          Bv.h[0] = *(const v8h*)vr;
          Bv.h[1] = *(const v8h*)(vr + 8);
          acc[t] = wmma16(Ap, Bv, acc[t]);
        }
      }
    }
  }

  // ---- epilogue: divide by running sum, store fp32 ----
#pragma unroll
  for (int r = 0; r < 8; ++r) {
    const float inv = 1.0f / li[r];
    const int row = q0 + r + 8 * hiw;
#pragma unroll
    for (int t = 0; t < 4; ++t)
      out[(size_t)(b * S_LEN + row) * D_HEAD + t * 16 + lm] = acc[t][r] * inv;
  }
}

extern "C" void kernel_launch(void* const* d_in, const int* in_sizes, int n_in,
                              void* d_out, int out_size, void* d_ws, size_t ws_size,
                              hipStream_t stream) {
  (void)in_sizes; (void)n_in; (void)out_size; (void)ws_size;
  const float* q  = (const float*)d_in[0];
  const float* k  = (const float*)d_in[1];
  const float* v  = (const float*)d_in[2];
  const float* WQ = (const float*)d_in[3];
  const float* bQ = (const float*)d_in[4];
  const float* WK = (const float*)d_in[5];
  const float* bK = (const float*)d_in[6];
  const float* WV = (const float*)d_in[7];
  const float* bV = (const float*)d_in[8];

  _Float16* qh  = (_Float16*)d_ws;                               // 2 MB
  _Float16* kh  = qh + (size_t)BATCH * S_LEN * D_HEAD;           // 2 MB
  _Float16* vhT = kh + (size_t)BATCH * S_LEN * D_HEAD;           // 2 MB

  // 3 projections x 1024 row-tiles, 8 waves per block -> 384 blocks
  proj_kernel<<<384, 256, 0, stream>>>(q, k, v, WQ, bQ, WK, bK, WV, bV,
                                       qh, kh, vhT);
  // 32 query tiles of 128 per batch -> 128 blocks, 8 waves each
  attn_kernel<<<128, 256, 0, stream>>>(qh, kh, vhT, (float*)d_out);
}
